// Linear1d_60790967108190
// MI455X (gfx1250) — matrix operand order repaired and checked by reading, not compiled, not run on hardware
//
#include <hip/hip_runtime.h>

typedef __bf16 bf16;
typedef __attribute__((ext_vector_type(4)))  float f32x4;
typedef __attribute__((ext_vector_type(8)))  float f32x8;
typedef __attribute__((ext_vector_type(4)))  bf16  bf16x4;
typedef __attribute__((ext_vector_type(8)))  bf16  bf16x8;
typedef __attribute__((ext_vector_type(16))) bf16  bf16x16;

#define K_TOT   512
#define N_TOT   256
#define KC      32                 // K per pipeline step == one WMMA K
#define NCHUNK  (K_TOT / KC)       // 16 steps
#define MBLK    64                 // rows per block
#define BPAD    40                 // padded LDS row stride in bf16 (32 data + 8 pad) -> 80B, conflict-free

__global__ __launch_bounds__(256)
void linear_wmma_bf16(const float* __restrict__ x,
                      const float* __restrict__ w,
                      const float* __restrict__ bias,
                      float* __restrict__ out)
{
    // 2*64*40*2 + 2*256*40*2 = 51200 bytes
    __shared__ __align__(16) bf16 Ab[2][MBLK][BPAD];
    __shared__ __align__(16) bf16 Bb[2][N_TOT][BPAD];

    const int t    = threadIdx.x;
    const int lane = t & 31;
    const int wv   = t >> 5;       // wave id 0..7
    const int mp   = wv >> 2;      // M-pair 0..1  -> M-tiles {2mp, 2mp+1} (rows 32*mp .. +31)
    const int nq   = wv & 3;       // N-quarter 0..3 -> cols nq*64 .. +63 (4 N-tiles)
    const int l15  = lane & 15;
    const int hi   = lane >> 4;    // 0 or 1

    const long mBase = (long)blockIdx.x * MBLK;
    const float* xblk = x + mBase * K_TOT;

    // register staging for the software pipeline
    f32x4 aS[2];
    f32x4 bS[8];

    auto load_chunk = [&](int kc) {
        const int koff = kc * KC + (t & 7) * 4;   // float index within row
#pragma unroll
        for (int i = 0; i < 2; ++i) {
            const int r = i * 32 + (t >> 3);      // 0..63
            aS[i] = __builtin_nontemporal_load(
                        (const f32x4*)(xblk + (long)r * K_TOT + koff));
        }
#pragma unroll
        for (int i = 0; i < 8; ++i) {
            const int r = i * 32 + (t >> 3);      // 0..255
            bS[i] = *(const f32x4*)(w + (long)r * K_TOT + koff);  // keep L2-temporal
        }
    };

    auto cvt4 = [](f32x4 v) {
        bf16x4 r;
        r[0] = (bf16)v[0]; r[1] = (bf16)v[1];
        r[2] = (bf16)v[2]; r[3] = (bf16)v[3];
        return r;
    };

    auto store_chunk = [&](int buf) {
        const int kk = (t & 7) * 4;               // bf16 index within padded row
#pragma unroll
        for (int i = 0; i < 2; ++i) {
            const int r = i * 32 + (t >> 3);
            *(bf16x4*)&Ab[buf][r][kk] = cvt4(aS[i]);
        }
#pragma unroll
        for (int i = 0; i < 8; ++i) {
            const int r = i * 32 + (t >> 3);
            *(bf16x4*)&Bb[buf][r][kk] = cvt4(bS[i]);
        }
    };

    // acc[2*j + mt]: N-tile j (0..3), M-subtile mt (0..1)
    f32x8 acc[8];
#pragma unroll
    for (int j = 0; j < 8; ++j)
#pragma unroll
        for (int v = 0; v < 8; ++v) acc[j][v] = 0.0f;

    // pipeline prologue
    load_chunk(0);
    store_chunk(0);
    __syncthreads();

    const int arow0 = mp * 32 + l15;       // A fragment rows for the two M-tiles
    const int arow1 = mp * 32 + 16 + l15;
    const int akb   = hi * 8;              // A: K base (split layout [8h,8h+8) U [8h+16,8h+24))
    const int bkb   = hi * 8;              // B: K base (split layout [8h,8h+8) U [8h+16,8h+24))

    auto loadB = [&](int buf, int j) {
        const int n = nq * 64 + j * 16 + l15;
        bf16x8 b0 = *(const bf16x8*)&Bb[buf][n][bkb];
        bf16x8 b1 = *(const bf16x8*)&Bb[buf][n][bkb + 16];
        return __builtin_shufflevector(b0, b1,
            0,1,2,3,4,5,6,7,8,9,10,11,12,13,14,15);
    };

#pragma unroll 2
    for (int kc = 0; kc < NCHUNK; ++kc) {
        const int cur = kc & 1;

        if (kc + 1 < NCHUNK) load_chunk(kc + 1);   // global loads overlap compute

        // Two A fragments (4x ds_load_b128, conflict-free)
        bf16x8 a00 = *(const bf16x8*)&Ab[cur][arow0][akb];
        bf16x8 a01 = *(const bf16x8*)&Ab[cur][arow0][akb + 16];
        bf16x8 a10 = *(const bf16x8*)&Ab[cur][arow1][akb];
        bf16x8 a11 = *(const bf16x8*)&Ab[cur][arow1][akb + 16];
        bf16x16 af0 = __builtin_shufflevector(a00, a01,
            0,1,2,3,4,5,6,7,8,9,10,11,12,13,14,15);
        bf16x16 af1 = __builtin_shufflevector(a10, a11,
            0,1,2,3,4,5,6,7,8,9,10,11,12,13,14,15);

        // B-fragment register double-buffer: next pair of ds_loads stays in
        // flight behind the two wmmas consuming the current fragment.
        bf16x16 bf_cur = loadB(cur, 0);
#pragma unroll
        for (int j = 0; j < 4; ++j) {
            bf16x16 bf_nxt;
            if (j < 3) bf_nxt = loadB(cur, j + 1);
            acc[2*j + 0] = __builtin_amdgcn_wmma_f32_16x16x32_bf16(
                false, af0, false, bf_cur, (short)0, acc[2*j + 0], false, false);
            acc[2*j + 1] = __builtin_amdgcn_wmma_f32_16x16x32_bf16(
                false, af1, false, bf_cur, (short)0, acc[2*j + 1], false, false);
            if (j < 3) bf_cur = bf_nxt;
        }

        if (kc + 1 < NCHUNK) store_chunk((kc + 1) & 1);
        __syncthreads();
    }

    // epilogue: D[v][lane]: row = v + 8*hi, col = l15 (within a 16x16 tile)
    const float scale = 0.125f;  // 1/sqrt(64)
#pragma unroll
    for (int j = 0; j < 4; ++j) {
        const int col = nq * 64 + j * 16 + l15;
        const float bv = 0.1f * bias[col];
#pragma unroll
        for (int mt = 0; mt < 2; ++mt) {
            const long m0 = mBase + mp * 32 + mt * 16 + hi * 8;
#pragma unroll
            for (int v = 0; v < 8; ++v) {
                __builtin_nontemporal_store(acc[2*j + mt][v] * scale + bv,
                                            &out[(m0 + v) * N_TOT + col]);
            }
        }
    }
}

extern "C" void kernel_launch(void* const* d_in, const int* in_sizes, int n_in,
                              void* d_out, int out_size, void* d_ws, size_t ws_size,
                              hipStream_t stream) {
    const float* x    = (const float*)d_in[0];  // [65536, 512]
    const float* wgt  = (const float*)d_in[1];  // [256, 512]
    const float* bias = (const float*)d_in[2];  // [1, 256]
    float* out = (float*)d_out;                 // [65536, 256]

    dim3 grid(65536 / MBLK);   // 1024 blocks
    dim3 block(256);           // 8 waves (wave32)
    linear_wmma_bf16<<<grid, block, 0, stream>>>(x, wgt, bias, out);
}
